// Attention_19481971654841
// MI455X (gfx1250) — matrix operand order repaired and checked by reading, not compiled, not run on hardware
//
#include <hip/hip_runtime.h>
#include <hip/hip_bf16.h>

// ---------------- problem constants ----------------
#define B_  2
#define L_  2048
#define D_  1024
#define H_  16
#define KVH_ 8
#define HD_ 128
#define NTOK (B_ * L_)        // 4096
#define QDIM (H_ * HD_)       // 2048
#define KVDIM (KVH_ * HD_)    // 1024
#define SCALE_ 0.08838834764831845f  // 1/sqrt(128)

typedef __attribute__((ext_vector_type(16))) __bf16 v16bf;
typedef __attribute__((ext_vector_type(8)))  float  v8f;

union Frag { v16bf v; unsigned short u[16]; };

__device__ inline unsigned short f2bf(float f) {
    unsigned int u = __float_as_uint(f);
    u += 0x7FFFu + ((u >> 16) & 1u);   // round-to-nearest-even
    return (unsigned short)(u >> 16);
}
__device__ inline float bf2f(unsigned short h) {
    return __uint_as_float(((unsigned int)h) << 16);
}

// load 4 consecutive elements as bf16 bit-patterns
__device__ inline void ld4_bf16(const float* p, unsigned short* o) {
    const float4 v = *(const float4*)p;
    o[0] = f2bf(v.x); o[1] = f2bf(v.y); o[2] = f2bf(v.z); o[3] = f2bf(v.w);
}
__device__ inline void ld4_bf16(const unsigned short* p, unsigned short* o) {
    const uint2 v = *(const uint2*)p;
    o[0] = (unsigned short)(v.x & 0xFFFFu); o[1] = (unsigned short)(v.x >> 16);
    o[2] = (unsigned short)(v.y & 0xFFFFu); o[3] = (unsigned short)(v.y >> 16);
}
__device__ inline void stC(float* p, float v) { *p = v; }
__device__ inline void stC(unsigned short* p, float v) { *p = f2bf(v); }

// ================= GEMM: C[M,N] = A[M,K] * Bw[N,K]^T (bf16 WMMA, f32 acc) ==========
#define BM 128
#define BN 128
#define BK 32
#define LDT 40   // LDS row stride in ushorts (pad for 16B-aligned frag reads)

template <typename TA, typename TC>
__global__ __launch_bounds__(256) void gemm_wmma_kernel(
    const TA* __restrict__ A, const float* __restrict__ Bw, TC* __restrict__ C,
    int M, int N, int K)
{
    __shared__ unsigned short As[BM * LDT];
    __shared__ unsigned short Bs[BN * LDT];
    const int tid = threadIdx.x;
    const int lane = tid & 31;
    const int w  = tid >> 5;
    const int wm = w & 3;        // 4 waves along M (32 rows each)
    const int wn = w >> 2;       // 2 waves along N (64 cols each)
    const int m0 = blockIdx.y * BM;
    const int n0 = blockIdx.x * BN;
    const int lh = lane >> 4;    // lane half (0/1)
    const int ll = lane & 15;

    v8f acc[2][4];
    for (int mt = 0; mt < 2; ++mt)
        for (int nt = 0; nt < 4; ++nt)
            for (int r = 0; r < 8; ++r) acc[mt][nt][r] = 0.0f;

    for (int kb = 0; kb < K; kb += BK) {
        __syncthreads();
        // cooperative stage: 128x32 A and 128x32 B tiles -> bf16 LDS
        for (int i = 0; i < 4; ++i) {
            int f  = tid + i * 256;       // float4-group id, 0..1023
            int r  = f >> 3;              // row within tile (0..127)
            int kq = (f & 7) << 2;        // k offset (0..28, step 4)
            unsigned short t4[4];
            ld4_bf16(&A[(size_t)(m0 + r) * K + kb + kq], t4);
            unsigned long long pk =
                (unsigned long long)t4[0] | ((unsigned long long)t4[1] << 16) |
                ((unsigned long long)t4[2] << 32) | ((unsigned long long)t4[3] << 48);
            *(unsigned long long*)&As[r * LDT + kq] = pk;
            ld4_bf16(&Bw[(size_t)(n0 + r) * K + kb + kq], t4);
            pk = (unsigned long long)t4[0] | ((unsigned long long)t4[1] << 16) |
                 ((unsigned long long)t4[2] << 32) | ((unsigned long long)t4[3] << 48);
            *(unsigned long long*)&Bs[r * LDT + kq] = pk;
            // pull next K-step tiles toward the caches (global_prefetch_b8)
            if (kb + BK < K) {
                __builtin_prefetch(&A[(size_t)(m0 + r) * K + kb + BK + kq], 0, 1);
                __builtin_prefetch(&Bw[(size_t)(n0 + r) * K + kb + BK + kq], 0, 1);
            }
        }
        __syncthreads();

        // A fragments: lane holds row m = base+ll; k = 8*lh + i + (i>=8 ? 8 : 0)
        Frag a[2], b[4];
        for (int mt = 0; mt < 2; ++mt) {
            int m  = wm * 32 + mt * 16 + ll;
            int kl = lh * 8;
            for (int j = 0; j < 4; ++j) {
                *(unsigned*)&a[mt].u[2 * j]     = *(const unsigned*)&As[m * LDT + kl + 2 * j];
                *(unsigned*)&a[mt].u[8 + 2 * j] = *(const unsigned*)&As[m * LDT + kl + 16 + 2 * j];
            }
        }
        // B fragments: lane holds col n = base+ll; k = 8*lh + i + (i>=8 ? 8 : 0)
        for (int nt = 0; nt < 4; ++nt) {
            int n  = wn * 64 + nt * 16 + ll;
            int ks = lh * 8;
            *(uint4*)&b[nt].u[0] = *(const uint4*)&Bs[n * LDT + ks];
            *(uint4*)&b[nt].u[8] = *(const uint4*)&Bs[n * LDT + ks + 16];
        }
        for (int mt = 0; mt < 2; ++mt)
            for (int nt = 0; nt < 4; ++nt)
                acc[mt][nt] = __builtin_amdgcn_wmma_f32_16x16x32_bf16(
                    false, a[mt].v, false, b[nt].v, (short)0, acc[mt][nt], false, false);
    }

    // D layout: element r -> row M = r + 8*lh, col N = ll
    for (int mt = 0; mt < 2; ++mt)
        for (int nt = 0; nt < 4; ++nt)
            for (int r = 0; r < 8; ++r) {
                int row = m0 + wm * 32 + mt * 16 + r + 8 * lh;
                int col = n0 + wn * 64 + nt * 16 + ll;
                stC(&C[(size_t)row * N + col], acc[mt][nt][r]);
            }
}

// ================= RMSNorm + RoPE + GQA relayout ==================================
// blockIdx: token*32 + unit; unit 0..15 = q heads, 16..23 = k heads, 24..31 = v heads
__global__ __launch_bounds__(64) void normrope_kernel(
    const unsigned short* __restrict__ q_raw,   // (NTOK, QDIM)
    const unsigned short* __restrict__ k_raw,   // (NTOK, KVDIM)
    const unsigned short* __restrict__ v_raw,   // (NTOK, KVDIM)
    const int* __restrict__ pos_ids,            // (B,L)
    const float* __restrict__ qw, const float* __restrict__ kw,
    unsigned short* __restrict__ q_fin,         // (B,H,L,HD)
    unsigned short* __restrict__ k_fin,         // (B,KVH,L,HD)
    unsigned short* __restrict__ v_fin)         // (B,KVH,L,HD)
{
    __shared__ float red[2];
    const int t = threadIdx.x;                  // 0..63, handles dims t and t+64
    const int unit  = blockIdx.x & 31;
    const int token = blockIdx.x >> 5;
    const int b = token / L_, l = token % L_;

    const unsigned short* src;
    unsigned short* dst;
    const float* nw = nullptr;
    if (unit < 16) {
        src = q_raw + (size_t)token * QDIM + unit * HD_;
        dst = q_fin + ((size_t)(b * H_ + unit) * L_ + l) * HD_;
        nw  = qw;
    } else if (unit < 24) {
        int kh = unit - 16;
        src = k_raw + (size_t)token * KVDIM + kh * HD_;
        dst = k_fin + ((size_t)(b * KVH_ + kh) * L_ + l) * HD_;
        nw  = kw;
    } else {
        int vh = unit - 24;                      // V: pure relayout
        src = v_raw + (size_t)token * KVDIM + vh * HD_;
        dst = v_fin + ((size_t)(b * KVH_ + vh) * L_ + l) * HD_;
        dst[t] = src[t]; dst[t + 64] = src[t + 64];
        return;                                  // uniform per block
    }

    float x1 = bf2f(src[t]), x2 = bf2f(src[t + 64]);
    float ss = x1 * x1 + x2 * x2;
    for (int m = 16; m >= 1; m >>= 1) ss += __shfl_xor(ss, m, 32);
    if ((t & 31) == 0) red[t >> 5] = ss;
    __syncthreads();
    float rms = rsqrtf((red[0] + red[1]) * (1.0f / 128.0f) + 1e-6f);
    float n1 = x1 * rms * nw[t];
    float n2 = x2 * rms * nw[t + 64];

    float pos = (float)pos_ids[token];
    float inv = __powf(1.0e6f, -(float)(2 * t) * (1.0f / 128.0f));
    float sn, cs;
    __sincosf(pos * inv, &sn, &cs);
    dst[t]      = f2bf(n1 * cs - n2 * sn);   // rotate_half: (x1,x2) -> x1*c - x2*s
    dst[t + 64] = f2bf(n2 * cs + n1 * sn);
}

// ================= Flash attention (causal, GQA groups=2) =========================
// block = 128 threads (4 waves); wave w handles 16 q rows; block = 64 q rows
__global__ __launch_bounds__(128) void flash_kernel(
    const unsigned short* __restrict__ Q,    // (B,H,L,HD)
    const unsigned short* __restrict__ Kt,   // (B,KVH,L,HD)
    const unsigned short* __restrict__ V,    // (B,KVH,L,HD)
    unsigned short* __restrict__ O)          // (B,L,H*HD)
{
    __shared__ unsigned short Vs[HD_ * 32];        // transposed V block [hd][key]
    __shared__ unsigned short Ps[4 * 16 * 32];     // per-wave P tile [m][key]

    const int tid = threadIdx.x, lane = tid & 31, w = tid >> 5;
    const int lh = lane >> 4, ll = lane & 15;
    const int bh = blockIdx.y;
    const int b = bh / H_, h = bh % H_;
    const int kvh = h >> 1;                        // GROUPS = 2
    const int qb = blockIdx.x;
    const int qbase = qb * 64 + w * 16;
    const size_t qoff = (size_t)(b * H_ + h) * L_ * HD_;
    const size_t koff = (size_t)(b * KVH_ + kvh) * L_ * HD_;

    // Q A-fragments for 4 HD chunks (loaded once)
    Frag qf[4];
    {
        int m  = qbase + ll;
        int kl = lh * 8;
        const unsigned short* qp = Q + qoff + (size_t)m * HD_;
        for (int c = 0; c < 4; ++c)
            for (int j = 0; j < 4; ++j) {
                *(unsigned*)&qf[c].u[2 * j]     = *(const unsigned*)(qp + c * 32 + kl + 2 * j);
                *(unsigned*)&qf[c].u[8 + 2 * j] = *(const unsigned*)(qp + c * 32 + kl + 16 + 2 * j);
            }
    }

    v8f acc[8];
    float mr[8], lr[8];
    for (int ht = 0; ht < 8; ++ht)
        for (int r = 0; r < 8; ++r) acc[ht][r] = 0.0f;
    for (int r = 0; r < 8; ++r) { mr[r] = -1e30f; lr[r] = 0.0f; }

    const int kbEnd = qb * 2 + 1;                   // causal: keys up to q block end
    for (int kb = 0; kb <= kbEnd; ++kb) {
        const int k0 = kb * 32;

        // prefetch next K/V block (one cacheline per lane; guarded in-bounds)
        if (kb < kbEnd) {
            const unsigned short* kn = Kt + koff + (size_t)(k0 + 32 + lane) * HD_;
            const unsigned short* vn = V  + koff + (size_t)(k0 + 32 + lane) * HD_;
            __builtin_prefetch(kn, 0, 1);
            __builtin_prefetch(vn, 0, 1);
        }

        __syncthreads();
        // stage V block transposed: Vs[hd][key]
        for (int i = 0; i < 4; ++i) {
            int c   = tid + i * 128;                // 0..511
            int key = c >> 4;
            int hc  = (c & 15) << 3;
            uint4 vv = *(const uint4*)(V + koff + (size_t)(k0 + key) * HD_ + hc);
            const unsigned short* pv = (const unsigned short*)&vv;
            for (int e = 0; e < 8; ++e) Vs[(hc + e) * 32 + key] = pv[e];
        }
        __syncthreads();

        // S = Q * K^T  (two 16x16 tiles over the 32-key block)
        v8f s[2];
        for (int t = 0; t < 2; ++t) {
            for (int r = 0; r < 8; ++r) s[t][r] = 0.0f;
            int key = k0 + t * 16 + ll;
            const unsigned short* kp = Kt + koff + (size_t)key * HD_;
            for (int c = 0; c < 4; ++c) {
                Frag kf;
                *(uint4*)&kf.u[0] = *(const uint4*)(kp + c * 32 + lh * 8);
                *(uint4*)&kf.u[8] = *(const uint4*)(kp + c * 32 + lh * 8 + 16);
                s[t] = __builtin_amdgcn_wmma_f32_16x16x32_bf16(
                    false, qf[c].v, false, kf.v, (short)0, s[t], false, false);
            }
        }

        // scale + causal mask + online softmax (rows: M = r + 8*lh, col = ll)
        float sc[2][8];
        for (int t = 0; t < 2; ++t)
            for (int r = 0; r < 8; ++r) {
                float v = s[t][r] * SCALE_;
                int key  = k0 + t * 16 + ll;
                int qrow = qbase + r + 8 * lh;
                sc[t][r] = (key > qrow) ? -1e30f : v;
            }
        for (int r = 0; r < 8; ++r) {
            float tm = fmaxf(sc[0][r], sc[1][r]);
            tm = fmaxf(tm, __shfl_xor(tm, 1, 16));
            tm = fmaxf(tm, __shfl_xor(tm, 2, 16));
            tm = fmaxf(tm, __shfl_xor(tm, 4, 16));
            tm = fmaxf(tm, __shfl_xor(tm, 8, 16));
            float mn = fmaxf(mr[r], tm);
            float al = __expf(mr[r] - mn);
            float rs = 0.0f;
            for (int t = 0; t < 2; ++t) { sc[t][r] = __expf(sc[t][r] - mn); rs += sc[t][r]; }
            rs += __shfl_xor(rs, 1, 16);
            rs += __shfl_xor(rs, 2, 16);
            rs += __shfl_xor(rs, 4, 16);
            rs += __shfl_xor(rs, 8, 16);
            lr[r] = lr[r] * al + rs;
            mr[r] = mn;
            for (int ht = 0; ht < 8; ++ht) acc[ht][r] *= al;
        }

        // P (D-layout) -> LDS -> A-layout fragment
        unsigned short* pw = &Ps[w * 16 * 32];
        for (int t = 0; t < 2; ++t)
            for (int r = 0; r < 8; ++r)
                pw[(r + 8 * lh) * 32 + t * 16 + ll] = f2bf(sc[t][r]);
        Frag pf;
        {
            int kl = lh * 8;
            for (int j = 0; j < 4; ++j) {
                *(unsigned*)&pf.u[2 * j]     = *(const unsigned*)&pw[ll * 32 + kl + 2 * j];
                *(unsigned*)&pf.u[8 + 2 * j] = *(const unsigned*)&pw[ll * 32 + kl + 16 + 2 * j];
            }
        }
        // acc += P * V  (8 HD tiles)
        for (int ht = 0; ht < 8; ++ht) {
            Frag vf;
            *(uint4*)&vf.u[0] = *(const uint4*)&Vs[(ht * 16 + ll) * 32 + lh * 8];
            *(uint4*)&vf.u[8] = *(const uint4*)&Vs[(ht * 16 + ll) * 32 + lh * 8 + 16];
            acc[ht] = __builtin_amdgcn_wmma_f32_16x16x32_bf16(
                false, pf.v, false, vf.v, (short)0, acc[ht], false, false);
        }
    }

    // normalize + store (B,L,H*HD)
    for (int ht = 0; ht < 8; ++ht)
        for (int r = 0; r < 8; ++r) {
            int qrow = qbase + r + 8 * lh;
            float ov = acc[ht][r] / lr[r];
            O[((size_t)(b * L_ + qrow)) * QDIM + h * HD_ + ht * 16 + ll] = f2bf(ov);
        }
}

// ================= launch ==========================================================
extern "C" void kernel_launch(void* const* d_in, const int* in_sizes, int n_in,
                              void* d_out, int out_size, void* d_ws, size_t ws_size,
                              hipStream_t stream)
{
    const float* hs  = (const float*)d_in[0];
    const int*   pos = (const int*)d_in[1];
    const float* qw  = (const float*)d_in[2];
    const float* kw  = (const float*)d_in[3];
    const float* vw  = (const float*)d_in[4];
    const float* ow  = (const float*)d_in[5];
    const float* qnw = (const float*)d_in[6];
    const float* knw = (const float*)d_in[7];
    float* out = (float*)d_out;

    unsigned short* q_raw = (unsigned short*)d_ws;
    unsigned short* k_raw = q_raw + (size_t)NTOK * QDIM;
    unsigned short* v_raw = k_raw + (size_t)NTOK * KVDIM;
    unsigned short* q_fin = v_raw + (size_t)NTOK * KVDIM;
    unsigned short* k_fin = q_fin + (size_t)NTOK * QDIM;
    unsigned short* v_fin = k_fin + (size_t)NTOK * KVDIM;
    unsigned short* attn  = v_fin + (size_t)NTOK * KVDIM;   // total ~80 MB

    dim3 blk(256);
    gemm_wmma_kernel<float, unsigned short>
        <<<dim3(QDIM / BN, NTOK / BM), blk, 0, stream>>>(hs, qw, q_raw, NTOK, QDIM, D_);
    gemm_wmma_kernel<float, unsigned short>
        <<<dim3(KVDIM / BN, NTOK / BM), blk, 0, stream>>>(hs, kw, k_raw, NTOK, KVDIM, D_);
    gemm_wmma_kernel<float, unsigned short>
        <<<dim3(KVDIM / BN, NTOK / BM), blk, 0, stream>>>(hs, vw, v_raw, NTOK, KVDIM, D_);

    normrope_kernel<<<NTOK * 32, 64, 0, stream>>>(q_raw, k_raw, v_raw, pos, qnw, knw,
                                                  q_fin, k_fin, v_fin);

    flash_kernel<<<dim3(L_ / 64, B_ * H_), 128, 0, stream>>>(q_fin, k_fin, v_fin, attn);

    gemm_wmma_kernel<unsigned short, float>
        <<<dim3(D_ / BN, NTOK / BM), blk, 0, stream>>>(attn, ow, out, NTOK, D_, QDIM);
}
